// FIVO_28544352650006
// MI455X (gfx1250) — compile-verified, matrix operand order checked by reading, not run on hardware
//
#include <hip/hip_runtime.h>

// ---------------------------------------------------------------------------
// FIVO particle filter for MI455X (gfx1250): one workgroup per batch element,
// full T-loop on-chip, WMMA bf16 GEMMs (64x128xK) fed from LDS.
// ---------------------------------------------------------------------------

typedef __attribute__((ext_vector_type(16))) __bf16 v16bf;
typedef __attribute__((ext_vector_type(8)))  float  v8f;

#define FIVO_B  256
#define FIVO_T  128
#define FIVO_P  64
#define FIVO_L  64
#define FIVO_E  64
#define FIVO_H  128
#define FIVO_D  2
#define FIVO_BIAS 0.5413248538970947f
#define FIVO_LOG2PI 1.8378770664093453f

union FragU { v16bf v; uint4 q[2]; };

__device__ __forceinline__ float softplusf(float v) {
  return v > 0.f ? v + log1pf(expf(-v)) : log1pf(expf(v));
}

__device__ __forceinline__ unsigned pcg_hash(unsigned v) {
  v = v * 747796405u + 2891336453u;
  unsigned w = ((v >> ((v >> 28u) + 4u)) ^ v) * 277803737u;
  return (w >> 22u) ^ w;
}
__device__ __forceinline__ float u01f(unsigned h) {
  return ((float)(h >> 8) + 0.5f) * (1.0f / 16777216.0f);
}
__device__ __forceinline__ float gaussf(unsigned s) {
  unsigned h1 = pcg_hash(s);
  unsigned h2 = pcg_hash(s ^ 0x9E3779B9u);
  float u1 = u01f(h1), u2 = u01f(h2);
  return sqrtf(-2.0f * logf(u1)) * cosf(6.28318530718f * u2);
}

// One wave accumulates a 16x16 tile: C += A[mtile*16..+15][0..K) * B^T rows.
// A: LDS bf16, row-major, lda elements per row.
// Bt: global bf16, [N][K] (pre-transposed), ldb = K elements per row.
// Fragment layout per CDNA5 ISA 7.12.2 (16-bit A/B: lane half selects 8-wide
// contiguous K chunks at +hi*8 and +16+hi*8).
__device__ __forceinline__ void wave_gemm_acc(v8f& acc,
                                              const __bf16* __restrict__ Alds, int lda,
                                              const __bf16* __restrict__ Bt, int ldb,
                                              int K, int mtile, int n0, int lane) {
  const int mrow = mtile * 16 + (lane & 15);
  const int ncol = n0 + (lane & 15);
  const int hi = lane >> 4;
  for (int kc = 0; kc < K; kc += 32) {
    FragU fa, fb;
    const __bf16* ap = Alds + mrow * lda + kc + hi * 8;
    fa.q[0] = *(const uint4*)(ap);
    fa.q[1] = *(const uint4*)(ap + 16);
    const __bf16* bp = Bt + ncol * ldb + kc + hi * 8;
    fb.q[0] = *(const uint4*)(bp);
    fb.q[1] = *(const uint4*)(bp + 16);
    acc = __builtin_amdgcn_wmma_f32_16x16x32_bf16(
        /*neg_a=*/false, fa.v, /*neg_b=*/false, fb.v,
        /*c_mod=*/(short)0, acc, /*reuse_a=*/false, /*reuse_b=*/false);
  }
}

// ---------------------------------------------------------------------------
// Prep: convert + transpose weights to bf16 [N][K] layout in workspace.
// ---------------------------------------------------------------------------
__global__ void fivo_prep(const float* __restrict__ W_prior,
                          const float* __restrict__ W_rx,
                          const float* __restrict__ W_rh,
                          const float* __restrict__ W_enc,
                          __bf16* __restrict__ WpzT,   // [128][64]
                          __bf16* __restrict__ WrxzT,  // [128][64]
                          __bf16* __restrict__ WrhT,   // [128][128]
                          __bf16* __restrict__ WencT)  // [128][128]
{
  int i = blockIdx.x * blockDim.x + threadIdx.x;
  if (i < 128 * 64) {
    int n = i >> 6, k = i & 63;
    WpzT[i]  = (__bf16)W_prior[(64 + k) * 128 + n];  // z rows of W_prior
    WrxzT[i] = (__bf16)W_rx[(64 + k) * 128 + n];     // z rows of W_rx
  }
  if (i < 128 * 128) {
    int n = i >> 7, k = i & 127;
    WrhT[i]  = (__bf16)W_rh[k * 128 + n];
    WencT[i] = (__bf16)W_enc[k * 128 + n];
  }
}

// ---------------------------------------------------------------------------
// Main: one block per batch element b; 256 threads = 8 wave32s.
// ---------------------------------------------------------------------------
__global__ __launch_bounds__(256) void fivo_main(
    const float* __restrict__ x, const float* __restrict__ y,
    const float* __restrict__ z0,
    const float* __restrict__ W_emb, const float* __restrict__ b_emb,
    const float* __restrict__ W_prior, const float* __restrict__ b_prior,
    const float* __restrict__ W_rx, const float* __restrict__ b_r,
    const float* __restrict__ b_enc,
    const float* __restrict__ W_dec, const float* __restrict__ b_dec,
    const __bf16* __restrict__ WpzT, const __bf16* __restrict__ WrxzT,
    const __bf16* __restrict__ WrhT, const __bf16* __restrict__ WencT,
    float* __restrict__ out)
{
  const int b = blockIdx.x;
  const int tid = threadIdx.x;
  const int lane = tid & 31;
  const int w = tid >> 5;
  const int mtile = w & 3;          // 4 row-tiles of 16 particles
  const int n0base = (w >> 2) * 64; // two wave-groups split N=128

  __shared__ __align__(16) __bf16 zA[FIVO_P][FIVO_L];   // current z
  __shared__ __align__(16) __bf16 zB[FIVO_P][FIVO_L];   // new z (pre-resample)
  __shared__ __align__(16) __bf16 rA[FIVO_P][FIVO_H];   // recurrent state
  __shared__ __align__(16) __bf16 FB[FIVO_P][FIVO_H];   // staging buffer
  __shared__ float exs[FIVO_E];
  __shared__ float pc[2 * FIVO_L];   // prior bias + x-part
  __shared__ float rc[FIVO_H];       // rnn bias + x,y-part
  __shared__ float bencS[2 * FIVO_L];
  __shared__ float lw[FIVO_P];       // log_w
  __shared__ float lwm[FIVO_P];      // log_w_m (8x8 flat)
  __shared__ float la[FIVO_P];       // log_alpha accum
  __shared__ float lpv[FIVO_P];
  __shared__ float cum[FIVO_P];
  __shared__ int   idxs[FIVO_P];
  __shared__ float scal[8];          // 0:log_p 1:log_p_m 2:lse 3:u_t 4:do_rs 5:cumtot

  // ---- init carry ----
  if (tid < FIVO_P) { lw[tid] = 0.f; lwm[tid] = 0.f; }
  if (tid < 8) scal[tid] = 0.f;
  if (tid < 2 * FIVO_L) bencS[tid] = b_enc[tid];
  for (int i = tid; i < FIVO_P * FIVO_L; i += 256) zA[i >> 6][i & 63] = (__bf16)z0[i & 63];
  for (int i = tid; i < FIVO_P * FIVO_H; i += 256) rA[i >> 7][i & 127] = (__bf16)0.0f;
  // warm the weight path into cache hierarchy (global_prefetch_b8)
  if (tid < 128) {
    __builtin_prefetch(WrhT + tid * 128, 0, 1);
    __builtin_prefetch(WencT + tid * 128, 0, 1);
    __builtin_prefetch(WpzT + tid * 64, 0, 1);
    __builtin_prefetch(WrxzT + tid * 64, 0, 1);
  }
  __syncthreads();

  for (int t = 0; t < FIVO_T; ++t) {
    const float x0 = x[(b * FIVO_T + t) * FIVO_D + 0];
    const float x1 = x[(b * FIVO_T + t) * FIVO_D + 1];
    const float y0 = y[(b * FIVO_T + t) * FIVO_D + 0];
    const float y1 = y[(b * FIVO_T + t) * FIVO_D + 1];

    // embed row (D=2 matvec) — particle independent
    if (tid < FIVO_E)
      exs[tid] = b_emb[tid] + x0 * W_emb[0 * FIVO_E + tid] + x1 * W_emb[1 * FIVO_E + tid];
    __syncthreads();

    // particle-independent bias vectors for prior / rnn
    if (tid < 2 * FIVO_L) {
      float s = b_prior[tid];
      float s2 = b_r[tid];
      for (int e = 0; e < FIVO_E; ++e) {
        s  += exs[e] * W_prior[e * (2 * FIVO_L) + tid];
        s2 += exs[e] * W_rx[e * FIVO_H + tid];
      }
      s2 += y0 * W_rx[(FIVO_E + FIVO_L + 0) * FIVO_H + tid];
      s2 += y1 * W_rx[(FIVO_E + FIVO_L + 1) * FIVO_H + tid];
      pc[tid] = s;
      rc[tid] = s2;
    }
    __syncthreads();

    // ---- RNN GEMM: tanh(z@Wrx_z + r@W_rh + rc) -> FB ----
    for (int nt = 0; nt < 4; ++nt) {
      const int n0 = n0base + nt * 16;
      v8f acc = {0.f, 0.f, 0.f, 0.f, 0.f, 0.f, 0.f, 0.f};
      wave_gemm_acc(acc, &zA[0][0], FIVO_L, WrxzT, FIVO_L, FIVO_L, mtile, n0, lane);
      wave_gemm_acc(acc, &rA[0][0], FIVO_H, WrhT, FIVO_H, FIVO_H, mtile, n0, lane);
      const int n = n0 + (lane & 15);
      const int hi = lane >> 4;
      const float bias = rc[n];
#pragma unroll
      for (int v = 0; v < 8; ++v) {
        const int m = mtile * 16 + hi * 8 + v;
        FB[m][n] = (__bf16)tanhf(acc[v] + bias);
      }
    }
    __syncthreads();
    for (int i = tid; i < FIVO_P * FIVO_H; i += 256) rA[i >> 7][i & 127] = FB[i >> 7][i & 127];
    __syncthreads();

    // ---- ENC GEMM: qp = r@W_enc + b_enc -> FB ----
    for (int nt = 0; nt < 4; ++nt) {
      const int n0 = n0base + nt * 16;
      v8f acc = {0.f, 0.f, 0.f, 0.f, 0.f, 0.f, 0.f, 0.f};
      wave_gemm_acc(acc, &rA[0][0], FIVO_H, WencT, FIVO_H, FIVO_H, mtile, n0, lane);
      const int n = n0 + (lane & 15);
      const int hi = lane >> 4;
      const float bias = bencS[n];
#pragma unroll
      for (int v = 0; v < 8; ++v) {
        const int m = mtile * 16 + hi * 8 + v;
        FB[m][n] = (__bf16)(acc[v] + bias);
      }
    }
    __syncthreads();

    // ---- reparameterized sample + posterior log-prob ----
    if (tid < FIVO_P) {
      const int m = tid;
      float acc_lp = 0.f;
      for (int j = 0; j < FIVO_L; ++j) {
        const float mu = (float)FB[m][j];
        const float sq = softplusf((float)FB[m][FIVO_L + j] + FIVO_BIAS);
        const unsigned seed = ((unsigned)t * 1315423911u) ^
                              ((unsigned)(b * FIVO_P + m) * 2654435761u) ^
                              ((unsigned)j * 40503u) ^ 0xA5117ACEu;
        const float eps = gaussf(seed);
        zB[m][j] = (__bf16)(mu + sq * eps);
        acc_lp += -0.5f * eps * eps - logf(sq);
      }
      const float post_lp = acc_lp - 0.5f * (float)FIVO_L * FIVO_LOG2PI;
      la[m] = -post_lp;
    }
    __syncthreads();

    // ---- PRIOR GEMM (reads OLD z in zA): pp = z@Wp_z + pc -> FB ----
    for (int nt = 0; nt < 4; ++nt) {
      const int n0 = n0base + nt * 16;
      v8f acc = {0.f, 0.f, 0.f, 0.f, 0.f, 0.f, 0.f, 0.f};
      wave_gemm_acc(acc, &zA[0][0], FIVO_L, WpzT, FIVO_L, FIVO_L, mtile, n0, lane);
      const int n = n0 + (lane & 15);
      const int hi = lane >> 4;
      const float bias = pc[n];
#pragma unroll
      for (int v = 0; v < 8; ++v) {
        const int m = mtile * 16 + hi * 8 + v;
        FB[m][n] = (__bf16)(acc[v] + bias);
      }
    }
    __syncthreads();

    // ---- prior log-prob + decoder data log-prob + weight update ----
    if (tid < FIVO_P) {
      const int m = tid;
      float plp = 0.f;
      float a0 = b_dec[0], a1 = b_dec[1], a2 = b_dec[2], a3 = b_dec[3];
      for (int j = 0; j < FIVO_L; ++j) {
        const float mu = (float)FB[m][j];
        const float sg = softplusf((float)FB[m][FIVO_L + j] + FIVO_BIAS);
        const float zv = (float)zB[m][j];
        const float d = (zv - mu) / sg;
        plp += -0.5f * d * d - logf(sg);
        a0 += zv * W_dec[j * 4 + 0];
        a1 += zv * W_dec[j * 4 + 1];
        a2 += zv * W_dec[j * 4 + 2];
        a3 += zv * W_dec[j * 4 + 3];
      }
      plp -= 0.5f * (float)FIVO_L * FIVO_LOG2PI;
      const float sg0 = softplusf(a1 + FIVO_BIAS);
      const float sg1 = softplusf(a3 + FIVO_BIAS);
      const float d0 = (y0 - a0) / sg0;
      const float d1 = (y1 - a2) / sg1;
      const float dlp = -0.5f * d0 * d0 - logf(sg0)
                        - 0.5f * d1 * d1 - logf(sg1)
                        - 0.5f * (float)FIVO_D * FIVO_LOG2PI;
      la[m] += plp + dlp;
      lpv[m] = lw[m] + la[m];
    }
    __syncthreads();

    // ---- per-b reductions: logsumexp, MIWAE term, u_t ----
    if (tid == 0) {
      float mx = -1e30f;
      for (int m = 0; m < FIVO_P; ++m) mx = fmaxf(mx, lpv[m]);
      float s = 0.f;
      for (int m = 0; m < FIVO_P; ++m) s += expf(lpv[m] - mx);
      const float lse = mx + logf(s);
      scal[0] += lse;
      scal[2] = lse;
      float accm = 0.f;
      for (int g = 0; g < 8; ++g) {
        float mg = -1e30f;
        for (int i = 0; i < 8; ++i) mg = fmaxf(mg, lwm[g * 8 + i] + la[g * 8 + i]);
        float sg = 0.f;
        for (int i = 0; i < 8; ++i) sg += expf(lwm[g * 8 + i] + la[g * 8 + i] - mg);
        accm += mg + logf(sg);
      }
      scal[1] += accm * 0.125f;
      const unsigned hu = pcg_hash(((unsigned)t * 2246822519u) ^ ((unsigned)b * 3266489917u) ^ 0x68BC21EBu);
      scal[3] = u01f(hu);
    }
    __syncthreads();
    if (tid < FIVO_P) lw[tid] = lpv[tid] - scal[2];
    __syncthreads();

    // ---- ESS + cumsum (serial, cheap) ----
    if (tid == 0) {
      float mx = -1e30f;
      for (int m = 0; m < FIVO_P; ++m) mx = fmaxf(mx, 2.f * lw[m]);
      float s = 0.f;
      for (int m = 0; m < FIVO_P; ++m) s += expf(2.f * lw[m] - mx);
      const float log_ess = -(mx + logf(s));
      scal[4] = (expf(log_ess) < (float)(FIVO_P / 2)) ? 1.f : 0.f;
      float c = 0.f;
      for (int m = 0; m < FIVO_P; ++m) { c += expf(lw[m]); cum[m] = c; }
      scal[5] = c;
    }
    __syncthreads();

    // ---- systematic resampling ----
    const bool do_rs = scal[4] > 0.5f;
    if (tid < FIVO_P) {
      const float tot = scal[5];
      const float pos = ((float)tid + scal[3]) / (float)FIVO_P;
      int id = 0;
      while (id < FIVO_P - 1 && (cum[id] / tot) < pos) ++id;  // searchsorted left, clipped
      idxs[tid] = do_rs ? id : tid;
      if (do_rs) lw[tid] = -logf((float)FIVO_P);
    }
    __syncthreads();
    // gather z_new -> zA (next step's z); renormalize group weights
    for (int i = tid; i < FIVO_P * FIVO_L; i += 256) {
      const int m = i >> 6, j = i & 63;
      zA[m][j] = zB[idxs[m]][j];
    }
    if (tid < 8) {
      float mg = -1e30f;
      for (int i = 0; i < 8; ++i) mg = fmaxf(mg, lw[tid * 8 + i]);
      float sg = 0.f;
      for (int i = 0; i < 8; ++i) sg += expf(lw[tid * 8 + i] - mg);
      const float l = mg + logf(sg);
      for (int i = 0; i < 8; ++i) lwm[tid * 8 + i] = lw[tid * 8 + i] - l;
    }
    __syncthreads();
  }

  if (tid == 0) out[b] = scal[0] + scal[1];
}

// ---------------------------------------------------------------------------
extern "C" void kernel_launch(void* const* d_in, const int* in_sizes, int n_in,
                              void* d_out, int out_size, void* d_ws, size_t ws_size,
                              hipStream_t stream) {
  const float* x       = (const float*)d_in[0];
  const float* y       = (const float*)d_in[1];
  const float* z0      = (const float*)d_in[2];
  const float* W_emb   = (const float*)d_in[3];
  const float* b_emb   = (const float*)d_in[4];
  const float* W_prior = (const float*)d_in[5];
  const float* b_prior = (const float*)d_in[6];
  const float* W_rx    = (const float*)d_in[7];
  const float* W_rh    = (const float*)d_in[8];
  const float* b_r     = (const float*)d_in[9];
  const float* W_enc   = (const float*)d_in[10];
  const float* b_enc   = (const float*)d_in[11];
  const float* W_dec   = (const float*)d_in[12];
  const float* b_dec   = (const float*)d_in[13];

  char* ws = (char*)d_ws;
  __bf16* WpzT  = (__bf16*)(ws + 0);       // 128*64*2  = 16 KB
  __bf16* WrxzT = (__bf16*)(ws + 16384);   // 128*64*2  = 16 KB
  __bf16* WrhT  = (__bf16*)(ws + 32768);   // 128*128*2 = 32 KB
  __bf16* WencT = (__bf16*)(ws + 65536);   // 128*128*2 = 32 KB

  fivo_prep<<<64, 256, 0, stream>>>(W_prior, W_rx, W_rh, W_enc,
                                    WpzT, WrxzT, WrhT, WencT);
  fivo_main<<<FIVO_B, 256, 0, stream>>>(x, y, z0, W_emb, b_emb, W_prior, b_prior,
                                        W_rx, b_r, b_enc, W_dec, b_dec,
                                        WpzT, WrxzT, WrhT, WencT, (float*)d_out);
}
